// SuperVLAD_70111046140645
// MI455X (gfx1250) — matrix operand order repaired and checked by reading, not compiled, not run on hardware
//
#include <hip/hip_runtime.h>
#include <hip/hip_bf16.h>

// ---------------------------------------------------------------------------
// SuperVLAD on MI455X (gfx1250): bf16 WMMA path, fused norm+logits+softmax.
// Memory-bound problem (~250MB moved vs 6.5 GFLOP) -> minimize DRAM traffic:
// convert x to normalized bf16 once, do both GEMMs with v_wmma_f32_16x16x32_bf16.
// ---------------------------------------------------------------------------

typedef __attribute__((ext_vector_type(16))) __bf16 v16bf;
typedef __attribute__((ext_vector_type(8)))  __bf16 v8bf;
typedef __attribute__((ext_vector_type(8)))  float  v8f;

#define D_DIM 768
#define T_DIM 1024
#define N_BATCH 32
#define K_CLUST 64           // clusters kept (ghost dropped)
#define KP 80                // padded cluster rows (5 x 16)
#define TT 128               // t-tile per block in k1
#define XPAD 776             // LDS row stride for 768 bf16 (+8 pad, kills bank conflicts)

static __device__ __forceinline__ v16bf cat8(v8bf lo, v8bf hi) {
  return __builtin_shufflevector(lo, hi, 0,1,2,3,4,5,6,7,8,9,10,11,12,13,14,15);
}

// ---------------------------------------------------------------------------
// k0: pad conv_weight (65x768 f32) to 80x768 bf16
// ---------------------------------------------------------------------------
__global__ __launch_bounds__(256) void k0_prep_w(const float* __restrict__ W,
                                                 __bf16* __restrict__ Wb) {
  int i = blockIdx.x * 256 + threadIdx.x;      // 0 .. 80*768-1 (grid = 240)
  int k = i / D_DIM, d = i % D_DIM;
  float v = (k < 65) ? W[k * D_DIM + d] : 0.0f;
  Wb[i] = (__bf16)v;
}

// ---------------------------------------------------------------------------
// k1: per (n, t-tile of 128): L2-normalize columns, cvt bf16 (LDS + ws),
//     logits GEMM (80x768 @ 768x128) with WMMA bf16, masked softmax over k,
//     store soft-assign bf16 for k<64.
// Dynamic LDS: xbT[128][776] bf16 + red[256] f32 + invn[128] f32 = 200192 B.
// ---------------------------------------------------------------------------
__global__ __launch_bounds__(256) void k1_norm_logits_softmax(
    const float*  __restrict__ x,     // (N, D, T)
    const __bf16* __restrict__ Wb,    // (80, 768)
    __bf16*       __restrict__ xn,    // (N, D, T) bf16 normalized
    __bf16*       __restrict__ soft)  // (N, 64, T) bf16
{
  extern __shared__ char smem[];
  __bf16* xb  = (__bf16*)smem;                         // [TT][XPAD] t-major
  float*  red = (float*)(smem + TT * XPAD * 2);        // 256
  float*  invn = red + 256;                            // 128

  const int n  = blockIdx.y;
  const int t0 = blockIdx.x * TT;
  const int tid = threadIdx.x;
  const int tl  = tid & 127;     // local t
  const int dh  = tid >> 7;      // 0/1

  const float* xp = x + ((size_t)n * D_DIM) * T_DIM + t0;

  // ---- pass 1: column sum-of-squares -------------------------------------
  float ss = 0.f;
  for (int d = dh; d < D_DIM; d += 2) {
    float v = xp[(size_t)d * T_DIM + tl];
    ss += v * v;
  }
  red[tid] = ss;
  __syncthreads();
  if (tid < 128) {
    float s = red[tid] + red[tid + 128];
    invn[tid] = 1.0f / fmaxf(sqrtf(s), 1e-12f);
  }
  __syncthreads();

  // ---- pass 2: normalize, cvt bf16, stage (LDS t-major) + spill to ws ----
  __bf16* xnp = xn + ((size_t)n * D_DIM) * T_DIM + t0;
  for (int d = dh; d < D_DIM; d += 2) {
    float v = xp[(size_t)d * T_DIM + tl] * invn[tl];
    __bf16 b = (__bf16)v;
    xb[tl * XPAD + d] = b;                 // transposed: row t, contiguous d
    xnp[(size_t)d * T_DIM + tl] = b;
  }
  __syncthreads();

  // ---- logits GEMM: wave w handles t-subtile w (16 t's), all 5 m-tiles ---
  const int wave = tid >> 5;
  const int lane = tid & 31;
  const int hl   = lane >> 4;    // half-lane select (K/M split per ISA layout)
  const int l16  = lane & 15;

  const v8f vzero = {0.f,0.f,0.f,0.f,0.f,0.f,0.f,0.f};
  v8f acc[5];
  #pragma unroll
  for (int m = 0; m < 5; ++m) acc[m] = vzero;

  const __bf16* xrow = xb + (wave * 16 + l16) * XPAD + hl * 16;  // B: N=t, K=d
  #pragma unroll 4
  for (int kk = 0; kk < D_DIM / 32; ++kk) {
    const __bf16* bp = xrow + kk * 32;
    v16bf bfrag = cat8(*(const v8bf*)bp, *(const v8bf*)(bp + 8));
    #pragma unroll
    for (int m = 0; m < 5; ++m) {
      const __bf16* ap = Wb + (m * 16 + l16) * D_DIM + kk * 32 + hl * 16;
      v16bf afrag = cat8(*(const v8bf*)ap, *(const v8bf*)(ap + 8));
      acc[m] = __builtin_amdgcn_wmma_f32_16x16x32_bf16(
                   false, afrag, false, bfrag, (short)0, acc[m], false, false);
    }
  }

  // ---- masked softmax over k (0..64 valid); M split across lane halves ---
  // lane holds k = m*16 + hl*8 + j for its t = t0 + wave*16 + l16
  float mx = -1e30f;
  #pragma unroll
  for (int m = 0; m < 4; ++m)
    #pragma unroll
    for (int j = 0; j < 8; ++j) mx = fmaxf(mx, acc[m][j]);
  if (hl == 0) mx = fmaxf(mx, acc[4][0]);          // ghost k=64
  mx = fmaxf(mx, __shfl_xor(mx, 16, 32));

  float se = 0.f;
  #pragma unroll
  for (int m = 0; m < 4; ++m)
    #pragma unroll
    for (int j = 0; j < 8; ++j) {
      float e = __expf(acc[m][j] - mx);
      acc[m][j] = e;
      se += e;
    }
  if (hl == 0) se += __expf(acc[4][0] - mx);       // ghost in denominator only
  se += __shfl_xor(se, 16, 32);
  float inv = 1.0f / se;

  const int t = t0 + wave * 16 + l16;
  #pragma unroll
  for (int m = 0; m < 4; ++m)
    #pragma unroll
    for (int j = 0; j < 8; ++j)
      soft[((size_t)n * K_CLUST + m * 16 + hl * 8 + j) * T_DIM + t] =
          (__bf16)(acc[m][j] * inv);
}

// ---------------------------------------------------------------------------
// k2: vlad GEMM per (n, d-tile of 128): C(64x128) += soft(64x1024) x xn^T.
//     xn is (d,t)-major so each lane loads its 16 contiguous t (k order shared with A) for B directly.
//     Writes raw vlad straight into d_out (n, k*768 + d).
// ---------------------------------------------------------------------------
__global__ __launch_bounds__(256) void k2_vlad(
    const __bf16* __restrict__ soft,  // (N, 64, T)
    const __bf16* __restrict__ xn,    // (N, D, T)
    float*        __restrict__ out)   // (N, 64*768)
{
  const int n   = blockIdx.y;
  const int tid = threadIdx.x;
  const int wave = tid >> 5;
  const int lane = tid & 31;
  const int hl   = lane >> 4;
  const int l16  = lane & 15;
  const int d0   = blockIdx.x * 128 + wave * 16 + l16;   // this lane's N (=d)

  const v8f vzero = {0.f,0.f,0.f,0.f,0.f,0.f,0.f,0.f};
  v8f acc[4];
  #pragma unroll
  for (int m = 0; m < 4; ++m) acc[m] = vzero;

  const __bf16* brow = xn + ((size_t)n * D_DIM + d0) * T_DIM + hl * 16;
  const __bf16* abase = soft + ((size_t)n * K_CLUST + l16) * T_DIM + hl * 16;

  #pragma unroll 4
  for (int tk = 0; tk < T_DIM / 32; ++tk) {
    const __bf16* bp = brow + tk * 32;
    __builtin_prefetch(bp + 128, 0, 1);               // global_prefetch_b8
    v16bf bfrag = cat8(*(const v8bf*)bp, *(const v8bf*)(bp + 8));
    #pragma unroll
    for (int m = 0; m < 4; ++m) {
      const __bf16* ap = abase + (size_t)(m * 16) * T_DIM + tk * 32;
      v16bf afrag = cat8(*(const v8bf*)ap, *(const v8bf*)(ap + 8));
      acc[m] = __builtin_amdgcn_wmma_f32_16x16x32_bf16(
                   false, afrag, false, bfrag, (short)0, acc[m], false, false);
    }
  }

  #pragma unroll
  for (int m = 0; m < 4; ++m)
    #pragma unroll
    for (int j = 0; j < 8; ++j)
      out[((size_t)n * K_CLUST + m * 16 + hl * 8 + j) * D_DIM + d0] = acc[m][j];
}

// ---------------------------------------------------------------------------
// k3: intra-normalize each (n,k) row of 768; record post-norm ssq per row.
// ---------------------------------------------------------------------------
__global__ __launch_bounds__(256) void k3_intranorm(float* __restrict__ out,
                                                    float* __restrict__ rowssq) {
  const int n = blockIdx.y, k = blockIdx.x;
  float* row = out + ((size_t)n * K_CLUST + k) * D_DIM;
  __shared__ float red[256];
  float ss = 0.f;
  for (int d = threadIdx.x; d < D_DIM; d += 256) { float v = row[d]; ss += v * v; }
  red[threadIdx.x] = ss;
  __syncthreads();
  for (int s = 128; s > 0; s >>= 1) {
    if (threadIdx.x < s) red[threadIdx.x] += red[threadIdx.x + s];
    __syncthreads();
  }
  float tot = red[0];
  float inv = 1.0f / fmaxf(sqrtf(tot), 1e-12f);
  for (int d = threadIdx.x; d < D_DIM; d += 256) row[d] *= inv;
  if (threadIdx.x == 0) rowssq[n * K_CLUST + k] = tot * inv * inv;
}

// k3b: deterministic per-n reduction (no float atomics -> replay-stable)
__global__ void k3b_reduce(const float* __restrict__ rowssq,
                           float* __restrict__ gssq) {
  int n = threadIdx.x;                 // blockDim = 32, grid = 1
  float s = 0.f;
  for (int k = 0; k < K_CLUST; ++k) s += rowssq[n * K_CLUST + k];
  gssq[n] = s;
}

// k4: global L2 normalize per batch, in place
__global__ __launch_bounds__(256) void k4_globalnorm(float* __restrict__ out,
                                                     const float* __restrict__ gssq) {
  size_t i = (size_t)blockIdx.x * 256 + threadIdx.x;
  int n = (int)(i / (K_CLUST * D_DIM));
  out[i] *= 1.0f / fmaxf(sqrtf(gssq[n]), 1e-12f);
}

// ---------------------------------------------------------------------------
// Workspace layout (bytes):
//   [0)        Wb    : 80*768 bf16          = 122880
//   [122880)   gssq  : 32 f32   (pad->128)
//   [123008)   rowssq: 32*64 f32            = 8192
//   [131328)   soft  : 32*64*1024 bf16      = 4194304
//   [4325632)  xn    : 32*768*1024 bf16     = 50331648
// Total ~54.7 MB.
// ---------------------------------------------------------------------------
extern "C" void kernel_launch(void* const* d_in, const int* in_sizes, int n_in,
                              void* d_out, int out_size, void* d_ws, size_t ws_size,
                              hipStream_t stream) {
  const float* x = (const float*)d_in[0];       // (32, 768, 32, 32)
  const float* W = (const float*)d_in[1];       // (65, 768)
  float* out = (float*)d_out;                   // (32, 49152)
  char* ws = (char*)d_ws;

  __bf16* Wb     = (__bf16*)(ws);
  float*  gssq   = (float*)(ws + 122880);
  float*  rowssq = (float*)(ws + 123008);
  __bf16* soft   = (__bf16*)(ws + 131328);
  __bf16* xn     = (__bf16*)(ws + 4325632);

  k0_prep_w<<<240, 256, 0, stream>>>(W, Wb);

  const size_t ldsK1 = (size_t)TT * XPAD * 2 + 256 * 4 + 128 * 4;  // 200192 B
  k1_norm_logits_softmax<<<dim3(T_DIM / TT, N_BATCH), 256, ldsK1, stream>>>(
      x, Wb, xn, soft);

  k2_vlad<<<dim3(D_DIM / 128, N_BATCH), 256, 0, stream>>>(soft, xn, out);

  k3_intranorm<<<dim3(K_CLUST, N_BATCH), 256, 0, stream>>>(out, rowssq);
  k3b_reduce<<<1, 32, 0, stream>>>(rowssq, gssq);
  k4_globalnorm<<<(N_BATCH * K_CLUST * D_DIM) / 256, 256, 0, stream>>>(out, gssq);
}
